// MutualMultiAttentionHead_38551626449020
// MI455X (gfx1250) — compile-verified, matrix operand order checked by reading, not run on hardware
//
#include <hip/hip_runtime.h>
#include <hip/hip_bf16.h>

typedef __attribute__((ext_vector_type(16))) _Float16 v16h;
typedef __attribute__((ext_vector_type(8)))  float    v8f;

#define NN 50000
#define NE 800000

// WMMA 16-bit fragment K-index for element i, lane-half h: K = i + 8*(i>=8) + 8*h
__device__ __forceinline__ int frag_k(int i, int h) {
    return (i & 7) + ((i >> 3) << 4) + 8 * h;
}

// ---------------------------------------------------------------------------
// Kernel 1: node projections  P = relu(X @ W + b),  X:[N,64]  W:[64,256]
// Block = 8 waves, covers 25 M-tiles (3125 = 125 * 25); weights staged ONCE
// per block in LDS in WMMA-fragment order (v16h per lane -> ds_load_b128).
// A fragments loaded directly from global as float4. Col-tiles processed in
// PAIRS (independent accumulators) so the second tile's WMMAs fill the first
// tile's WMMA->VALU hazard slots; epilogue results go to distinct registers
// before a clause of immediate-offset stores. gridDim.z selects weight set.
// ---------------------------------------------------------------------------
__global__ void node_gemm_kernel(const float* __restrict__ X,
                                 const float* __restrict__ Wq, const float* __restrict__ bq, float* __restrict__ Pq,
                                 const float* __restrict__ Wk, const float* __restrict__ bk, float* __restrict__ Pk,
                                 const float* __restrict__ Wv, const float* __restrict__ bv, float* __restrict__ Pv) {
    __shared__ v16h  BsF[16][2][32];   // [col-tile][k-step][lane] fragments, 32 KB
    __shared__ float bias_s[256];

    const float* W; const float* bias; float* P;
    if (blockIdx.z == 0)      { W = Wq; bias = bq; P = Pq; }
    else if (blockIdx.z == 1) { W = Wk; bias = bk; P = Pk; }
    else                      { W = Wv; bias = bv; P = Pv; }

    const int tid = threadIdx.x;            // 256 threads

    // Stage weight fragments: 16 tiles x 2 k-steps x 32 lanes = 1024 fragments
    for (int p = tid; p < 1024; p += 256) {
        const int ct = p >> 6, ks = (p >> 5) & 1, lp = p & 31;
        const int mm = lp & 15, hh = lp >> 4;
        const int n  = ct * 16 + mm;
        v16h h;
        #pragma unroll
        for (int i = 0; i < 16; ++i) {
            const int kk = ks * 32 + frag_k(i, hh);
            h[i] = (_Float16)W[kk * 256 + n];
        }
        BsF[ct][ks][lp] = h;
    }
    bias_s[tid] = bias[tid];
    __syncthreads();

    const int wave = tid >> 5, lane = tid & 31;
    const int half = lane >> 4, m = lane & 15;

    for (int it = wave; it < 25; it += 8) {      // wave-uniform tile loop
        const int row0 = (blockIdx.x * 25 + it) * 16;

        // A fragments (2 k-steps) directly from global: 4 float4 loads each
        const float4* xr = (const float4*)(X + (size_t)(row0 + m) * 64);
        v16h a[2];
        #pragma unroll
        for (int ks = 0; ks < 2; ++ks) {
            #pragma unroll
            for (int q = 0; q < 4; ++q) {
                const int kk0 = ks * 32 + frag_k(q * 4, half);  // contiguous quad
                float4 v = xr[kk0 >> 2];
                a[ks][q * 4 + 0] = (_Float16)v.x;
                a[ks][q * 4 + 1] = (_Float16)v.y;
                a[ks][q * 4 + 2] = (_Float16)v.z;
                a[ks][q * 4 + 3] = (_Float16)v.w;
            }
        }

        // Single store base: element (r, ct) lives at bp[r*256 + ct*16]
        float* const bp = P + (size_t)(row0 + 8 * half) * 256 + m;

        #pragma unroll
        for (int ct = 0; ct < 16; ct += 2) {     // process col-tiles in pairs
            v8f acc0 = {}, acc1 = {};
            acc0 = __builtin_amdgcn_wmma_f32_16x16x32_f16(
                false, a[0], false, BsF[ct][0][lane], (short)0, acc0, false, false);
            acc0 = __builtin_amdgcn_wmma_f32_16x16x32_f16(
                false, a[1], false, BsF[ct][1][lane], (short)0, acc0, false, false);
            acc1 = __builtin_amdgcn_wmma_f32_16x16x32_f16(
                false, a[0], false, BsF[ct + 1][0][lane], (short)0, acc1, false, false);
            acc1 = __builtin_amdgcn_wmma_f32_16x16x32_f16(
                false, a[1], false, BsF[ct + 1][1][lane], (short)0, acc1, false, false);

            const float b0 = bias_s[ct * 16 + m];
            const float b1 = bias_s[ct * 16 + 16 + m];
            float res0[8], res1[8];
            #pragma unroll
            for (int r = 0; r < 8; ++r) {        // C: reg r -> row r + 8*half
                float v0 = acc0[r] + b0;
                float v1 = acc1[r] + b1;
                res0[r] = v0 > 0.f ? v0 : 0.f;
                res1[r] = v1 > 0.f ? v1 : 0.f;
            }
            #pragma unroll
            for (int r = 0; r < 8; ++r) {
                bp[r * 256 + ct * 16]      = res0[r];
                bp[r * 256 + ct * 16 + 16] = res1[r];
            }
        }
    }
}

// ---------------------------------------------------------------------------
// Kernel 2: per-edge qe/ke projection + sigmoid + edge attention -> colw[E,4]
// plus segment sums: deg[dst]+=1, wqsum[dst,h]+=wq[h], wksum[dst,h]+=wk[h]
// ---------------------------------------------------------------------------
__global__ void edge_attn_kernel(const float* __restrict__ Ein, const int* __restrict__ dst,
                                 const float* __restrict__ qew, const float* __restrict__ qeb,
                                 const float* __restrict__ kew, const float* __restrict__ keb,
                                 float* __restrict__ colw, float* __restrict__ wqsum,
                                 float* __restrict__ wksum, float* __restrict__ deg) {
    __shared__ float sqw[128], skw[128], sqb[4], skb[4];
    const int tid = threadIdx.x;
    if (tid < 128) { sqw[tid] = qew[tid]; skw[tid] = kew[tid]; }
    if (tid < 4)   { sqb[tid] = qeb[tid]; skb[tid] = keb[tid]; }
    __syncthreads();

    const int e = blockIdx.x * 256 + tid;   // E % 256 == 0

    float x[32];
    #pragma unroll
    for (int i = 0; i < 8; ++i) {
        float4 v = ((const float4*)(Ein + (size_t)e * 32))[i];
        x[4*i] = v.x; x[4*i+1] = v.y; x[4*i+2] = v.z; x[4*i+3] = v.w;
    }

    float wq[4], wk[4];
    #pragma unroll
    for (int h = 0; h < 4; ++h) {
        float aq = sqb[h], ak = skb[h];
        #pragma unroll
        for (int c = 0; c < 32; ++c) {
            aq += x[c] * sqw[c * 4 + h];
            ak += x[c] * skw[c * 4 + h];
        }
        wq[h] = 1.f / (1.f + __expf(-aq));
        wk[h] = 1.f / (1.f + __expf(-ak));
    }

    // s[h][g] = min(exp(wq[h]*wk[g]/sqrt(32)), 5); softmax over g; colw = mean_h
    const float inv_scale = 0.17677669529663687f;  // 1/sqrt(32)
    float cw[4] = {0.f, 0.f, 0.f, 0.f};
    #pragma unroll
    for (int h = 0; h < 4; ++h) {
        float s[4], mx = -1e30f;
        #pragma unroll
        for (int g = 0; g < 4; ++g) {
            float v = fminf(__expf(wq[h] * wk[g] * inv_scale), 5.f);
            s[g] = v; mx = fmaxf(mx, v);
        }
        float t[4], sum = 0.f;
        #pragma unroll
        for (int g = 0; g < 4; ++g) { t[g] = __expf(s[g] - mx); sum += t[g]; }
        const float r = 0.25f / sum;
        #pragma unroll
        for (int g = 0; g < 4; ++g) cw[g] += t[g] * r;
    }
    #pragma unroll
    for (int g = 0; g < 4; ++g) colw[(size_t)e * 4 + g] = cw[g];

    const int d = dst[e];
    atomicAdd(&deg[d], 1.0f);
    #pragma unroll
    for (int h = 0; h < 4; ++h) {
        atomicAdd(&wqsum[(size_t)d * 4 + h], wq[h]);
        atomicAdd(&wksum[(size_t)d * 4 + h], wk[h]);
    }
}

// ---------------------------------------------------------------------------
// Kernel 3: fused edge value GEMM + attention combine.
// Block = 8 waves; ve weights staged once per block in fragment-order LDS
// (8 KB). Each wave processes 10 edge-tiles of 16 edges (625*8*10 = 50000).
// edge_v = relu(Ein @ Wv + bv) via 8 WMMAs (K=32) issued in pairs, folded
// immediately with colw into edge_emb[16,32] -> d_out (no intermediate).
// ---------------------------------------------------------------------------
__global__ void edge_emb_kernel(const float* __restrict__ Ein,
                                const float* __restrict__ Wv, const float* __restrict__ bv,
                                const float* __restrict__ colw, float* __restrict__ outE) {
    __shared__ v16h  BsF[8][32];     // [col-tile][lane] fragments, 8 KB
    __shared__ float bias_s[128];

    const int tid = threadIdx.x;
    {   // 8 tiles x 32 lanes = 256 fragments: exactly one per thread
        const int ct = tid >> 5, lp = tid & 31;
        const int mm = lp & 15, hh = lp >> 4;
        const int n  = ct * 16 + mm;
        v16h h;
        #pragma unroll
        for (int i = 0; i < 16; ++i)
            h[i] = (_Float16)Wv[frag_k(i, hh) * 128 + n];
        BsF[ct][lp] = h;
    }
    if (tid < 128) bias_s[tid] = bv[tid];
    __syncthreads();

    const int wave = tid >> 5, lane = tid & 31;
    const int half = lane >> 4, m = lane & 15;

    for (int it = 0; it < 10; ++it) {
        const int e0 = (blockIdx.x * 80 + it * 8 + wave) * 16;

        // A fragment directly from global: 4 float4 loads (contiguous quads)
        const float4* er = (const float4*)(Ein + (size_t)(e0 + m) * 32);
        v16h a;
        #pragma unroll
        for (int q = 0; q < 4; ++q) {
            float4 v = er[frag_k(q * 4, half) >> 2];
            a[q * 4 + 0] = (_Float16)v.x;
            a[q * 4 + 1] = (_Float16)v.y;
            a[q * 4 + 2] = (_Float16)v.z;
            a[q * 4 + 3] = (_Float16)v.w;
        }

        // Attention column weights: rows r+8*half; all 16 lanes of a half
        // read the same float4 -> hardware broadcast from one cacheline.
        float cwv[8][4];
        #pragma unroll
        for (int r = 0; r < 8; ++r) {
            float4 w4 = *(const float4*)(colw + (size_t)(e0 + r + 8 * half) * 4);
            cwv[r][0] = w4.x; cwv[r][1] = w4.y; cwv[r][2] = w4.z; cwv[r][3] = w4.w;
        }

        float emblo[8] = {}, embhi[8] = {};
        #pragma unroll
        for (int ct = 0; ct < 8; ct += 2) {  // tile pair: cols g*32+d, d<16 / d>=16
            v8f acc0 = {}, acc1 = {};
            acc0 = __builtin_amdgcn_wmma_f32_16x16x32_f16(
                false, a, false, BsF[ct][lane], (short)0, acc0, false, false);
            acc1 = __builtin_amdgcn_wmma_f32_16x16x32_f16(
                false, a, false, BsF[ct + 1][lane], (short)0, acc1, false, false);

            const int   g  = ct >> 1;
            const float b0 = bias_s[ct * 16 + m];
            const float b1 = bias_s[ct * 16 + 16 + m];
            #pragma unroll
            for (int r = 0; r < 8; ++r) {
                float v0 = acc0[r] + b0;
                float v1 = acc1[r] + b1;
                v0 = v0 > 0.f ? v0 : 0.f;
                v1 = v1 > 0.f ? v1 : 0.f;
                const float w = cwv[r][g];
                emblo[r] += w * v0;
                embhi[r] += w * v1;
            }
        }

        // Single store base: row r lives at bp[r*32] / bp[r*32+16]
        float* const bp = outE + (size_t)(e0 + 8 * half) * 32 + m;
        #pragma unroll
        for (int r = 0; r < 8; ++r) {
            bp[r * 32]      = emblo[r];
            bp[r * 32 + 16] = embhi[r];
        }
    }
}

// ---------------------------------------------------------------------------
// Kernel 4: scatter  aggq[dst] += Pq[src], aggk[dst] += Pk[src]  (256 f32/edge)
// One wave per edge, 8 floats per lane, f32 global atomics.
// ---------------------------------------------------------------------------
__global__ void scatter_kernel(const float* __restrict__ Pq, const float* __restrict__ Pk,
                               const int* __restrict__ src, const int* __restrict__ dst,
                               float* __restrict__ aggq, float* __restrict__ aggk) {
    const int tid  = threadIdx.x;
    const int lane = tid & 31;
    const int e    = blockIdx.x * 8 + (tid >> 5);
    const int s = src[e], d = dst[e];

    const float4* q4 = (const float4*)(Pq + (size_t)s * 256) + lane * 2;
    const float4* k4 = (const float4*)(Pk + (size_t)s * 256) + lane * 2;
    __builtin_prefetch(k4, 0, 3);           // global_prefetch_b8 for the k gather

    float4 q0 = q4[0], q1 = q4[1];
    float* aq = aggq + (size_t)d * 256 + lane * 8;
    atomicAdd(aq + 0, q0.x); atomicAdd(aq + 1, q0.y);
    atomicAdd(aq + 2, q0.z); atomicAdd(aq + 3, q0.w);
    atomicAdd(aq + 4, q1.x); atomicAdd(aq + 5, q1.y);
    atomicAdd(aq + 6, q1.z); atomicAdd(aq + 7, q1.w);

    float4 k0 = k4[0], k1 = k4[1];
    float* ak = aggk + (size_t)d * 256 + lane * 8;
    atomicAdd(ak + 0, k0.x); atomicAdd(ak + 1, k0.y);
    atomicAdd(ak + 2, k0.z); atomicAdd(ak + 3, k0.w);
    atomicAdd(ak + 4, k1.x); atomicAdd(ak + 5, k1.y);
    atomicAdd(ak + 6, k1.z); atomicAdd(ak + 7, k1.w);
}

// ---------------------------------------------------------------------------
// Kernel 5: node finalize. One wave per node: mean-reduce, 4x4 attention
// (cross-lane shuffle reductions over DN=64), combine with node_v.
// ---------------------------------------------------------------------------
__global__ void node_final_kernel(const float* __restrict__ aggq, const float* __restrict__ aggk,
                                  const float* __restrict__ Pv,
                                  const float* __restrict__ wqsum, const float* __restrict__ wksum,
                                  const float* __restrict__ deg, float* __restrict__ outN) {
    const int tid  = threadIdx.x;
    const int lane = tid & 31;
    const int n    = blockIdx.x * 8 + (tid >> 5);

    const float dg   = deg[n];
    const float rdeg = dg > 0.f ? 1.0f / dg : 0.f;   // where(deg>0, /max(deg,1), 0)

    float q[4][2], k[4][2];
    #pragma unroll
    for (int h = 0; h < 4; ++h) {
        const float wq = wqsum[(size_t)n * 4 + h];
        const float wk = wksum[(size_t)n * 4 + h];
        #pragma unroll
        for (int j = 0; j < 2; ++j) {
            const int d = lane + 32 * j;
            q[h][j] = (aggq[(size_t)n * 256 + h * 64 + d] + wq) * rdeg;
            k[h][j] = (aggk[(size_t)n * 256 + h * 64 + d] + wk) * rdeg;
        }
    }

    float sc[4][4];
    #pragma unroll
    for (int h = 0; h < 4; ++h)
        #pragma unroll
        for (int g = 0; g < 4; ++g) {
            float p = q[h][0] * k[g][0] + q[h][1] * k[g][1];
            #pragma unroll
            for (int off = 16; off > 0; off >>= 1)
                p += __shfl_xor(p, off, 32);
            sc[h][g] = p;
        }

    const float inv_scale = 0.125f;          // 1/sqrt(64)
    float cw[4] = {0.f, 0.f, 0.f, 0.f};
    #pragma unroll
    for (int h = 0; h < 4; ++h) {
        float s[4], mx = -1e30f;
        #pragma unroll
        for (int g = 0; g < 4; ++g) {
            float v = fminf(__expf(sc[h][g] * inv_scale), 5.f);
            s[g] = v; mx = fmaxf(mx, v);
        }
        float t[4], sum = 0.f;
        #pragma unroll
        for (int g = 0; g < 4; ++g) { t[g] = __expf(s[g] - mx); sum += t[g]; }
        const float r = 0.25f / sum;
        #pragma unroll
        for (int g = 0; g < 4; ++g) cw[g] += t[g] * r;
    }

    #pragma unroll
    for (int j = 0; j < 2; ++j) {
        const int d = lane + 32 * j;
        float acc = 0.f;
        #pragma unroll
        for (int g = 0; g < 4; ++g)
            acc += cw[g] * Pv[(size_t)n * 256 + g * 64 + d];
        outN[(size_t)n * 64 + d] = acc;
    }
}

// ---------------------------------------------------------------------------
extern "C" void kernel_launch(void* const* d_in, const int* in_sizes, int n_in,
                              void* d_out, int out_size, void* d_ws, size_t ws_size,
                              hipStream_t stream) {
    const float* node_in = (const float*)d_in[0];
    const float* edge_in = (const float*)d_in[1];
    const int*   src     = (const int*)d_in[2];
    const int*   dst     = (const int*)d_in[3];
    const float* qn_w = (const float*)d_in[4],  *qn_b = (const float*)d_in[5];
    const float* qe_w = (const float*)d_in[6],  *qe_b = (const float*)d_in[7];
    const float* kn_w = (const float*)d_in[8],  *kn_b = (const float*)d_in[9];
    const float* ke_w = (const float*)d_in[10], *ke_b = (const float*)d_in[11];
    const float* vn_w = (const float*)d_in[12], *vn_b = (const float*)d_in[13];
    const float* ve_w = (const float*)d_in[14], *ve_b = (const float*)d_in[15];

    const size_t NP = (size_t)NN * 256;
    float* ws    = (float*)d_ws;
    float* aggq  = ws;                       // [N,256]
    float* aggk  = aggq  + NP;               // [N,256]
    float* wqsum = aggk  + NP;               // [N,4]
    float* wksum = wqsum + (size_t)NN * 4;   // [N,4]
    float* deg   = wksum + (size_t)NN * 4;   // [N]
    float* pq    = deg   + NN;               // [N,256]
    float* pk    = pq    + NP;               // [N,256]
    float* pv    = pk    + NP;               // [N,256]
    float* colw  = pv    + NP;               // [E,4]

    float* out_node = (float*)d_out;                       // [N,64]
    float* out_edge = out_node + (size_t)NN * 64;          // [E,32]

    // Zero accumulators (aggq, aggk, wqsum, wksum, deg are contiguous)
    hipMemsetAsync(ws, 0, (2 * NP + (size_t)9 * NN) * sizeof(float), stream);

    node_gemm_kernel<<<dim3(125, 1, 3), 256, 0, stream>>>(
        node_in, qn_w, qn_b, pq, kn_w, kn_b, pk, vn_w, vn_b, pv);

    edge_attn_kernel<<<NE / 256, 256, 0, stream>>>(
        edge_in, dst, qe_w, qe_b, ke_w, ke_b, colw, wqsum, wksum, deg);

    edge_emb_kernel<<<625, 256, 0, stream>>>(
        edge_in, ve_w, ve_b, colw, out_edge);

    scatter_kernel<<<NE / 8, 256, 0, stream>>>(pq, pk, src, dst, aggq, aggk);

    node_final_kernel<<<NN / 8, 256, 0, stream>>>(
        aggq, aggk, pv, wqsum, wksum, deg, out_node);
}
